// MultiHeadAttention_78245714199296
// MI455X (gfx1250) — matrix operand order repaired and checked by reading, not compiled, not run on hardware
//
#include <hip/hip_runtime.h>

typedef unsigned short u16;
typedef unsigned int   u32;
typedef __attribute__((ext_vector_type(16))) __bf16 v16bf;
typedef __attribute__((ext_vector_type(8)))  float  v8f;
typedef __attribute__((ext_vector_type(4)))  u32    v4u;
typedef __attribute__((ext_vector_type(8)))  int    v8i;
typedef __attribute__((ext_vector_type(4)))  int    v4i;

#define D_MODEL 1024
#define NHEADS  16
#define DK      64
#define SEQ     2048
#define BATCH   4
#define MROWS   (BATCH * SEQ)   // 8192

#if defined(__has_builtin)
#  if __has_builtin(__builtin_amdgcn_tensor_load_to_lds) && __has_builtin(__builtin_amdgcn_s_wait_tensorcnt)
#    define USE_TDM 1
#  endif
#endif
#ifndef USE_TDM
#  define USE_TDM 0
#endif

union FragU { uint4 u[2]; v16bf v; };

__device__ __forceinline__ u16 f2bf(float f) {
    unsigned u = __float_as_uint(f);
    u += 0x7FFFu + ((u >> 16) & 1u);       // round-to-nearest-even
    return (u16)(u >> 16);
}

// Assemble a 32-byte WMMA fragment from two 16-byte chunks.
__device__ __forceinline__ v16bf mkfrag(const u16* p0, const u16* p1) {
    FragU f;
    f.u[0] = *(const uint4*)p0;
    f.u[1] = *(const uint4*)p1;
    return f.v;
}

// Generic shared pointer -> wave-relative LDS byte address (ISA 10.2: addr[31:0]).
__device__ __forceinline__ unsigned lds_addr32(const void* p) {
    return (unsigned)(unsigned long long)p;
}

// CDNA5 async global->LDS (ASYNCcnt-tracked), GV addressing mode.
__device__ __forceinline__ void async_ld_b128(unsigned lds, const void* g) {
    asm volatile("global_load_async_to_lds_b128 %0, %1, off"
                 :: "v"(lds), "v"(g) : "memory");
}
__device__ __forceinline__ void wait_async0() {
    asm volatile("s_wait_asynccnt 0x0" ::: "memory");
}

#if USE_TDM
// Tensor Data Mover: DMA a [rows x 32] bf16 tile (row pitch 1024 elems) from
// global into LDS, padding 4 DWORDs after every 16 DWORDs (64B row -> 80B row)
// to match the [rows][40] padded LDS layout. One issue per wave; TENSORcnt.
__device__ __forceinline__ void tdm_load_tile(const u16* gsrc, unsigned lds, int rows) {
    unsigned long long ga = (unsigned long long)gsrc;
    v4u g0;
    g0[0] = 1u;                                          // count=1 (valid user D#)
    g0[1] = lds;                                         // lds_addr
    g0[2] = (u32)(ga & 0xFFFFFFFFu);                     // global_addr[31:0]
    g0[3] = ((u32)(ga >> 32) & 0x01FFFFFFu) | 0x80000000u; // global_addr[56:32] | type=2
    v8i g1;
    //        data_size=2B     pad_en      pad_interval=16DW  pad_amount=4DW
    g1[0] = (int)((1u << 16) | (1u << 20) | (3u << 22) | (3u << 25));
    g1[1] = (int)(32u << 16);          // tensor_dim0[15:0]=32 in bits[63:48]
    g1[2] = (int)((u32)rows << 16);    // tensor_dim0 hi=0; tensor_dim1 lo=rows
    g1[3] = (int)(32u << 16);          // tensor_dim1 hi=0; tile_dim0=32
    g1[4] = rows;                      // tile_dim1=rows; tile_dim2=0
    g1[5] = D_MODEL;                   // tensor_dim0_stride lo32 = 1024
    g1[6] = 0;                         // stride hi; tensor_dim1_stride lo (unused, 2D)
    g1[7] = 0;
    v4i z4 = {0, 0, 0, 0};
    v8i z8 = {0, 0, 0, 0, 0, 0, 0, 0};
    // 6-arg form (clang-23 / therock HIP headers): groups 0..3 + extra + cpol
    __builtin_amdgcn_tensor_load_to_lds(g0, g1, z4, z4, z8, 0);
}
#endif

// ---------------------------------------------------------------------------
// fp32 -> bf16 conversion (one pass; halves all downstream GEMM read traffic)
// ---------------------------------------------------------------------------
__global__ void cvt_f32_bf16(const float* __restrict__ src, u16* __restrict__ dst, int n) {
    int i = blockIdx.x * blockDim.x + threadIdx.x;
    int stride = gridDim.x * blockDim.x;
    for (; i < n; i += stride) dst[i] = f2bf(src[i]);
}

// ---------------------------------------------------------------------------
// C[M=8192, N=1024] = X * W^T, bf16 inputs, WMMA 16x16x32 bf16 -> f32.
// Workgroup tile 64(M) x 128(N), 4 waves (2x2), each wave 32x64 (2x4 subtiles).
// TDM path: double-buffered LDS, wave 0 DMAs the next K-slice while all waves
// run WMMAs on the current one.
// OUTMODE 0: bf16 into head-major [b,h,n,64] (projections); 1: fp32 flat.
// ---------------------------------------------------------------------------
template <int OUTMODE>
__global__ __launch_bounds__(128) void gemm_xwT(const u16* __restrict__ X,
                                                const u16* __restrict__ W,
                                                void* __restrict__ outp) {
    __shared__ __align__(16) u16 As[2][64][40];    // 2 x (64 x 32 + pad)
    __shared__ __align__(16) u16 Bs[2][128][40];   // 2 x (128 x 32 + pad)

    const int tid  = threadIdx.x;
    const int lane = tid & 31;
    const int wave = tid >> 5;
    const int m0 = blockIdx.y * 64;
    const int n0 = blockIdx.x * 128;
    const int wr = (wave >> 1) * 32;
    const int wc = (wave & 1) * 64;
    const int kh8  = (lane >> 4) << 3;
    const int kh16 = (lane >> 4) << 4;

    v8f zero = {};
    v8f acc[2][4];
    for (int i = 0; i < 2; i++)
        for (int j = 0; j < 4; j++) acc[i][j] = zero;

#if USE_TDM
    if (wave == 0) {
        tdm_load_tile(X + (size_t)m0 * D_MODEL, lds_addr32(&As[0][0][0]), 64);
        tdm_load_tile(W + (size_t)n0 * D_MODEL, lds_addr32(&Bs[0][0][0]), 128);
        __builtin_amdgcn_s_wait_tensorcnt(0);
    }
    __syncthreads();
#endif

    for (int k0 = 0; k0 < D_MODEL; k0 += 32) {
#if USE_TDM
        const int cur = (k0 >> 5) & 1;
        if ((k0 + 32 < D_MODEL) && wave == 0) {   // prefetch next slice via TDM
            tdm_load_tile(X + (size_t)m0 * D_MODEL + (k0 + 32),
                          lds_addr32(&As[cur ^ 1][0][0]), 64);
            tdm_load_tile(W + (size_t)n0 * D_MODEL + (k0 + 32),
                          lds_addr32(&Bs[cur ^ 1][0][0]), 128);
        }
#else
        const int cur = 0;
        __syncthreads();
        for (int i = 0; i < 2; i++) {
            int c = tid + i * 128;
            int r = c >> 2, cc = (c & 3) << 3;
            *(uint4*)&As[0][r][cc] = *(const uint4*)&X[(size_t)(m0 + r) * D_MODEL + k0 + cc];
        }
        for (int i = 0; i < 4; i++) {
            int c = tid + i * 128;
            int r = c >> 2, cc = (c & 3) << 3;
            *(uint4*)&Bs[0][r][cc] = *(const uint4*)&W[(size_t)(n0 + r) * D_MODEL + k0 + cc];
        }
        __syncthreads();
#endif

        v16bf af[2], bfr[4];
        for (int mi = 0; mi < 2; mi++) {
            const u16* p = &As[cur][wr + mi * 16 + (lane & 15)][kh8];
            af[mi] = mkfrag(p, p + 16);
        }
        for (int ni = 0; ni < 4; ni++) {
            const u16* p = &Bs[cur][wc + ni * 16 + (lane & 15)][kh8];
            bfr[ni] = mkfrag(p, p + 16);
        }
        for (int mi = 0; mi < 2; mi++)
            for (int ni = 0; ni < 4; ni++)
                acc[mi][ni] = __builtin_amdgcn_wmma_f32_16x16x32_bf16(
                    false, af[mi], false, bfr[ni], (short)0, acc[mi][ni], false, false);

#if USE_TDM
        if (k0 + 32 < D_MODEL) {
            if (wave == 0) __builtin_amdgcn_s_wait_tensorcnt(0);
            __syncthreads();
        }
#endif
    }

    for (int mi = 0; mi < 2; mi++)
        for (int ni = 0; ni < 4; ni++)
            for (int r = 0; r < 8; r++) {
                int m = m0 + wr + mi * 16 + r + ((lane >> 4) << 3);
                int n = n0 + wc + ni * 16 + (lane & 15);
                float val = acc[mi][ni][r];
                if (OUTMODE == 0) {
                    u16* out = (u16*)outp;
                    int b = m >> 11, nn = m & (SEQ - 1);
                    int h = n >> 6,  d  = n & (DK - 1);
                    out[(((size_t)((b << 4) | h) * SEQ + nn) << 6) | d] = f2bf(val);
                } else {
                    ((float*)outp)[(size_t)m * D_MODEL + n] = val;
                }
            }
}

// ---------------------------------------------------------------------------
// Flash attention, causal. Grid: (32 query tiles of 64, 64 batch*heads).
// Block 128 = 4 waves; wave handles 16 query rows, K/V blocks of 64 keys.
// K block fetched with CDNA5 async global->LDS; V transposed on store.
// ---------------------------------------------------------------------------
__global__ __launch_bounds__(128) void flash_attn(const u16* __restrict__ q,
                                                  const u16* __restrict__ k,
                                                  const u16* __restrict__ v,
                                                  u16* __restrict__ ctx) {
    __shared__ __align__(16) u16 Ks[64][72];       // K block, row-major [j][d]
    __shared__ __align__(16) u16 Vt[64][72];       // V block, transposed [d][j]
    __shared__ __align__(16) u16 Ps[4][16][72];    // per-wave P staging [row][j]

    const int tid = threadIdx.x, lane = tid & 31, wave = tid >> 5;
    const int bh = blockIdx.y, qt = blockIdx.x;
    const int b = bh >> 4, h = bh & 15;
    const size_t base = (size_t)bh * SEQ * DK;
    const int q0 = qt * 64 + wave * 16;
    const int kh8 = (lane >> 4) << 3, kh16 = (lane >> 4) << 4;

    v16bf aq[2];
    {
        const u16* qrow = q + base + (size_t)(q0 + (lane & 15)) * DK;
        for (int kc = 0; kc < 2; kc++) {
            const u16* p = qrow + kc * 32 + kh8;
            aq[kc] = mkfrag(p, p + 16);
        }
    }

    v8f zero = {};
    v8f oacc[4];
    for (int i = 0; i < 4; i++) oacc[i] = zero;
    float mrun[8], lrun[8];
    for (int r = 0; r < 8; r++) { mrun[r] = -3.0e38f; lrun[r] = 0.0f; }

    for (int kb = 0; kb <= qt; kb++) {
        __syncthreads();
        // K block: 512 16B-chunks via async global->LDS (no VGPR round-trip)
        for (int i = 0; i < 4; i++) {
            int c = tid + i * 128;
            int j = c >> 3, cc = (c & 7) << 3;
            async_ld_b128(lds_addr32(&Ks[j][cc]),
                          &k[base + (size_t)(kb * 64 + j) * DK + cc]);
        }
        // V block: load contiguous in d, store transposed [d][j]
        for (int i = 0; i < 4; i++) {
            int c = tid + i * 128;
            int j = c >> 3, d0 = (c & 7) << 3;
            uint4 raw = *(const uint4*)&v[base + (size_t)(kb * 64 + j) * DK + d0];
            const u16* e = (const u16*)&raw;
            for (int t = 0; t < 8; t++) Vt[d0 + t][j] = e[t];
        }
        if (kb < qt)
            __builtin_prefetch(&k[base + (size_t)((kb + 1) * 64) * DK], 0, 3);
        wait_async0();
        __syncthreads();

        // S = Q * K^T (contract over d); scale + causal mask
        float sb[4][8];
        for (int ni = 0; ni < 4; ni++) {
            v8f s = zero;
            for (int kc = 0; kc < 2; kc++) {
                const u16* p = &Ks[ni * 16 + (lane & 15)][kc * 32 + kh8];
                v16bf bk = mkfrag(p, p + 16);
                s = __builtin_amdgcn_wmma_f32_16x16x32_bf16(
                        false, aq[kc], false, bk, (short)0, s, false, false);
            }
            int jg = kb * 64 + ni * 16 + (lane & 15);
            for (int r = 0; r < 8; r++) {
                float val = s[r] * 0.125f;   // 1/sqrt(64)
                if (kb == qt) {
                    int mg = q0 + r + ((lane >> 4) << 3);
                    if (jg > mg) val = -3.0e38f;
                }
                sb[ni][r] = val;
            }
        }

        // Online softmax: rows live across 16 lanes of a half-wave
        for (int r = 0; r < 8; r++) {
            float rmax = fmaxf(fmaxf(sb[0][r], sb[1][r]), fmaxf(sb[2][r], sb[3][r]));
            for (int msk = 1; msk < 16; msk <<= 1)
                rmax = fmaxf(rmax, __shfl_xor(rmax, msk, 32));
            float mnew  = fmaxf(mrun[r], rmax);
            float alpha = __expf(mrun[r] - mnew);
            float psum = 0.0f;
            for (int ni = 0; ni < 4; ni++) {
                float p = __expf(sb[ni][r] - mnew);
                sb[ni][r] = p;
                psum += p;
            }
            for (int msk = 1; msk < 16; msk <<= 1)
                psum += __shfl_xor(psum, msk, 32);
            lrun[r] = lrun[r] * alpha + psum;
            mrun[r] = mnew;
            for (int ni = 0; ni < 4; ni++) oacc[ni][r] *= alpha;
        }

        // P: C-layout -> A-fragment via wave-private LDS (DS in-order per wave)
        for (int ni = 0; ni < 4; ni++)
            for (int r = 0; r < 8; r++)
                Ps[wave][r + ((lane >> 4) << 3)][ni * 16 + (lane & 15)] = f2bf(sb[ni][r]);

        v16bf ap[2];
        for (int kc = 0; kc < 2; kc++) {
            const u16* p = &Ps[wave][lane & 15][kc * 32 + kh8];
            ap[kc] = mkfrag(p, p + 16);
        }
        for (int ni = 0; ni < 4; ni++)
            for (int kc = 0; kc < 2; kc++) {
                const u16* p = &Vt[ni * 16 + (lane & 15)][kc * 32 + kh8];
                v16bf bv = mkfrag(p, p + 16);
                oacc[ni] = __builtin_amdgcn_wmma_f32_16x16x32_bf16(
                               false, ap[kc], false, bv, (short)0, oacc[ni], false, false);
            }
    }

    for (int r = 0; r < 8; r++) {
        float inv = 1.0f / lrun[r];
        int m = q0 + r + ((lane >> 4) << 3);
        size_t rowbase = (size_t)(b * SEQ + m) * D_MODEL + h * DK;
        for (int ni = 0; ni < 4; ni++)
            ctx[rowbase + ni * 16 + (lane & 15)] = f2bf(oacc[ni][r] * inv);
    }
}

// ---------------------------------------------------------------------------
extern "C" void kernel_launch(void* const* d_in, const int* in_sizes, int n_in,
                              void* d_out, int out_size, void* d_ws, size_t ws_size,
                              hipStream_t stream) {
    const float* Q  = (const float*)d_in[0];
    const float* K  = (const float*)d_in[1];
    const float* V  = (const float*)d_in[2];
    const float* wq = (const float*)d_in[3];
    const float* wk = (const float*)d_in[4];
    const float* wv = (const float*)d_in[5];
    const float* wo = (const float*)d_in[6];

    char* ws = (char*)d_ws;
    const size_t SZ_X = (size_t)MROWS * D_MODEL * sizeof(u16);     // 16 MiB
    const size_t SZ_W = (size_t)D_MODEL * D_MODEL * sizeof(u16);   //  2 MiB
    u16* Qb  = (u16*)(ws);
    u16* Kb  = (u16*)(ws + SZ_X);
    u16* Vb  = (u16*)(ws + 2 * SZ_X);
    u16* Wq  = (u16*)(ws + 3 * SZ_X);
    u16* Wk  = (u16*)(ws + 3 * SZ_X + SZ_W);
    u16* Wv  = (u16*)(ws + 3 * SZ_X + 2 * SZ_W);
    u16* Wo  = (u16*)(ws + 3 * SZ_X + 3 * SZ_W);
    u16* qh  = (u16*)(ws + 3 * SZ_X + 4 * SZ_W);
    u16* kh  = (u16*)(ws + 4 * SZ_X + 4 * SZ_W);
    u16* vh  = (u16*)(ws + 5 * SZ_X + 4 * SZ_W);
    u16* ctx = (u16*)(ws + 6 * SZ_X + 4 * SZ_W);
    // total workspace use: 7*16 MiB + 4*2 MiB = 120 MiB

    const int nX = MROWS * D_MODEL;
    const int nW = D_MODEL * D_MODEL;
    cvt_f32_bf16<<<1024, 256, 0, stream>>>(Q,  Qb, nX);
    cvt_f32_bf16<<<1024, 256, 0, stream>>>(K,  Kb, nX);
    cvt_f32_bf16<<<1024, 256, 0, stream>>>(V,  Vb, nX);
    cvt_f32_bf16<<<512,  256, 0, stream>>>(wq, Wq, nW);
    cvt_f32_bf16<<<512,  256, 0, stream>>>(wk, Wk, nW);
    cvt_f32_bf16<<<512,  256, 0, stream>>>(wv, Wv, nW);
    cvt_f32_bf16<<<512,  256, 0, stream>>>(wo, Wo, nW);

    dim3 gg(D_MODEL / 128, MROWS / 64);   // (8, 128)
    dim3 gb(128);
    gemm_xwT<0><<<gg, gb, 0, stream>>>(Qb, Wq, qh);
    gemm_xwT<0><<<gg, gb, 0, stream>>>(Kb, Wk, kh);
    gemm_xwT<0><<<gg, gb, 0, stream>>>(Vb, Wv, vh);

    flash_attn<<<dim3(SEQ / 64, BATCH * NHEADS), gb, 0, stream>>>(qh, kh, vh, ctx);

    gemm_xwT<1><<<gg, gb, 0, stream>>>(ctx, Wo, d_out);
}
